// PillarBackBone8x_origin_27401891348909
// MI455X (gfx1250) — matrix operand order repaired and checked by reading, not compiled, not run on hardware
//
#include <hip/hip_runtime.h>
#include <hip/hip_bf16.h>

// ---------------------------------------------------------------------------
// PillarBackBone8x on gfx1250 (MI455X): implicit-GEMM conv via v_wmma_f32_16x16x32_f16
// Activations staged f16 NHWC; weights f16 [Cout][9*Cin]; f32 accumulate;
// fused BN + ReLU + sparsity-mask epilogue. CIN templated so all per-K-step
// address math strength-reduces to shifts; K-loop software-pipelined.
// ---------------------------------------------------------------------------

typedef __attribute__((ext_vector_type(16))) _Float16 v16h;
typedef __attribute__((ext_vector_type(8)))  float    v8f;

#define GRID_H 432
#define GRID_W 496

// ---------------- utility kernels ----------------

__global__ void k_zero16(_Float16* p, int n8) {
  int i = blockIdx.x * blockDim.x + threadIdx.x;
  if (i < n8) ((uint4*)p)[i] = make_uint4(0u, 0u, 0u, 0u);
}

// scatter pillar features into f16 NHWC x0 (C=32) + f16 mask
__global__ void k_scatter(const float* __restrict__ feats,
                          const int*   __restrict__ coords,
                          _Float16* __restrict__ x0,
                          _Float16* __restrict__ m0, int N) {
  int i = blockIdx.x * blockDim.x + threadIdx.x;
  if (i >= N * 32) return;
  int n = i >> 5, c = i & 31;
  int b = coords[n * 3 + 0];
  int y = coords[n * 3 + 1];
  int x = coords[n * 3 + 2];
  int p = (b * GRID_H + y) * GRID_W + x;
  x0[p * 32 + c] = (_Float16)feats[n * 32 + c];
  if (c == 0) m0[p] = (_Float16)1.0f;
}

// weights HWIO f32 (k-major rows of Cout) -> f16 [Cout][K], K = 9*Cin
__global__ void k_wt(const float* __restrict__ w, _Float16* __restrict__ wt,
                     int K, int Cout) {
  int i = blockIdx.x * blockDim.x + threadIdx.x;
  if (i >= K * Cout) return;
  int co = i / K;
  int k  = i - co * K;
  wt[i] = (_Float16)w[k * Cout + co];
}

// fold BN: scale = g*rsqrt(v+eps), bias = b - m*scale
__global__ void k_bn(const float* __restrict__ g, const float* __restrict__ b,
                     const float* __restrict__ m, const float* __restrict__ v,
                     float* __restrict__ sc, float* __restrict__ bi, int C) {
  int i = blockIdx.x * blockDim.x + threadIdx.x;
  if (i >= C) return;
  float s = g[i] * rsqrtf(v[i] + 1e-3f);
  sc[i] = s;
  bi[i] = b[i] - m[i] * s;
}

// 3x3 max-pool, stride 2, pad 1 (mask dilation)
__global__ void k_dilate(const _Float16* __restrict__ mi, _Float16* __restrict__ mo,
                         int Bn, int Hin, int Win, int Hout, int Wout) {
  int i = blockIdx.x * blockDim.x + threadIdx.x;
  int tot = Bn * Hout * Wout;
  if (i >= tot) return;
  int hw = Hout * Wout;
  int b = i / hw;
  int r = i - b * hw;
  int yo = r / Wout;
  int xo = r - yo * Wout;
  float mx = 0.0f;
  for (int dy = 0; dy < 3; ++dy)
    for (int dx = 0; dx < 3; ++dx) {
      int yi = 2 * yo + dy - 1;
      int xi = 2 * xo + dx - 1;
      if ((unsigned)yi < (unsigned)Hin && (unsigned)xi < (unsigned)Win)
        mx = fmaxf(mx, (float)mi[(b * Hin + yi) * Win + xi]);
    }
  mo[i] = (_Float16)mx;
}

// ---------------- WMMA implicit-GEMM conv3x3, pad 1 ----------------
// M = Bn*Hout*Wout, N = Cout, K = 9*CIN. 256 threads = 8 waves computing a
// 64x32 C tile (4 waves in M, 2 in N); grid.y = Cout/32. Software-pipelined:
// global loads for K-step k+1 issue while step k is fed to WMMA from LDS.
template <int CIN>
__global__ __launch_bounds__(256)
void k_conv_wmma(const _Float16* __restrict__ in, const _Float16* __restrict__ wt,
                 const float* __restrict__ sc, const float* __restrict__ bi,
                 const _Float16* __restrict__ mask,
                 _Float16* __restrict__ out16, float* __restrict__ out32,
                 int Bn, int Hin, int Win, int Hout, int Wout,
                 int Cout, int stride) {
  constexpr int K      = 9 * CIN;
  constexpr int KSTEPS = K / 32;

  __shared__ __align__(16) _Float16 At[64 * 40]; // 64 pixels x 32 k (pad 8)
  __shared__ __align__(16) _Float16 Bt[32 * 40]; // 32 cout  x 32 k (pad 8)

  const int HWo = Hout * Wout;
  const int Mt  = Bn * HWo;

  const int tid  = threadIdx.x;
  const int lane = tid & 31;
  const int wv   = tid >> 5;
  const int wm   = wv & 3;   // wave M index (0..3)
  const int wn   = wv >> 2;  // wave N index (0..1)
  const int grp  = lane >> 4;
  const int lr   = lane & 15;

  const int pixBase = blockIdx.x * 64;
  const int nBase   = blockIdx.y * 32;

  // ---- per-thread staging coordinates (fixed across K loop) ----
  const int arow = tid >> 2;        // A: pixel row in tile, 0..63
  const int acol = (tid & 3) * 8;   // A: k-chunk, {0,8,16,24}
  const int apix = pixBase + arow;
  const bool avalid = apix < Mt;
  int ab = 0, ay = 0, ax = 0;
  if (avalid) {
    ab = apix / HWo;
    int r = apix - ab * HWo;
    ay = r / Wout;
    ax = r - ay * Wout;
  }
  const int ybase = ay * stride - 1;
  const int xbase = ax * stride - 1;
  const _Float16* __restrict__ inB = in + ab * Hin * Win * CIN;

  const int bnr = (tid >> 2) & 31;  // B: cout row 0..31 (tid < 128 active)
  const int bcol = (tid & 3) * 8;
  const _Float16* __restrict__ wrow = wt + (nBase + bnr) * K;

  // compile-time-constant divisions -> shifts/mads
  auto loadA = [&](int kk) -> uint4 {
    uint4 v = make_uint4(0u, 0u, 0u, 0u);
    const int k0  = kk * 32;
    const int tap = k0 / CIN;         // 3x3 tap index 0..8
    const int ciB = k0 - tap * CIN;   // channel base within tap
    const int ky  = tap / 3;
    const int kx  = tap - ky * 3;
    if (avalid) {
      int yi = ybase + ky;
      int xi = xbase + kx;
      if ((unsigned)yi < (unsigned)Hin && (unsigned)xi < (unsigned)Win)
        v = *(const uint4*)(inB + (yi * Win + xi) * CIN + ciB + acol);
    }
    return v;
  };
  auto loadB = [&](int kk) -> uint4 {
    return *(const uint4*)(wrow + kk * 32 + bcol);
  };

  v8f acc = {};

  // ---- software-pipelined K loop ----
  uint4 aR = loadA(0);
  uint4 bR = make_uint4(0u, 0u, 0u, 0u);
  if (tid < 128) bR = loadB(0);

  for (int kk = 0; kk < KSTEPS; ++kk) {
    *(uint4*)&At[arow * 40 + acol] = aR;
    if (tid < 128) *(uint4*)&Bt[bnr * 40 + bcol] = bR;
    __syncthreads();

    if (kk + 1 < KSTEPS) {            // prefetch next step during compute
      aR = loadA(kk + 1);
      if (tid < 128) bR = loadB(kk + 1);
    }

    // Fragments per CDNA5 16-bit WMMA VGPR layouts:
    // A: lane holds row m=lane&15; halfs j<8 -> K=grp*8+j, j>=8 -> K=16+grp*8+(j-8)
    // B: lane holds col n=lane&15; halfs j<8 -> K=grp*8+j, j>=8 -> K=16+grp*8+(j-8)
    union { v16h v; uint4 q[2]; } Af, Bf;
    const uint4* ar = (const uint4*)&At[(wm * 16 + lr) * 40];
    Af.q[0] = ar[grp];
    Af.q[1] = ar[2 + grp];
    const uint4* br = (const uint4*)&Bt[(wn * 16 + lr) * 40];
    Bf.q[0] = br[grp];
    Bf.q[1] = br[2 + grp];

    acc = __builtin_amdgcn_wmma_f32_16x16x32_f16(
        /*neg_a=*/false, Af.v, /*neg_b=*/false, Bf.v,
        /*c_mod=*/(short)0, acc, /*reuse_a=*/false, /*reuse_b=*/false);

    __syncthreads();
  }

  // Epilogue: C layout VGPR r -> (M = r + 8*grp, N = lane&15). Fuse BN+ReLU+mask.
  const int   ch = nBase + wn * 16 + lr;
  const float s  = sc[ch];
  const float bb = bi[ch];
#pragma unroll
  for (int r = 0; r < 8; ++r) {
    int pix = pixBase + wm * 16 + grp * 8 + r;
    if (pix < Mt) {
      float v = fmaxf(s * acc[r] + bb, 0.0f);
      if (mask) v *= (float)mask[pix];
      if (out32) { // final layer: f32 NCHW straight to d_out
        int b  = pix / HWo;
        int rm = pix - b * HWo;
        int yo = rm / Wout;
        int xo = rm - yo * Wout;
        out32[((b * Cout + ch) * Hout + yo) * Wout + xo] = v;
      } else {     // intermediate: f16 NHWC
        out16[pix * Cout + ch] = (_Float16)v;
      }
    }
  }
}

// ---------------- host orchestration ----------------

extern "C" void kernel_launch(void* const* d_in, const int* in_sizes, int n_in,
                              void* d_out, int out_size, void* d_ws, size_t ws_size,
                              hipStream_t stream) {
  const float* feats  = (const float*)d_in[0];
  const int*   coords = (const int*)d_in[1];
  const int N  = in_sizes[0] / 32; // 60000 pillars
  const int Bn = 2;

  // ---- workspace carve-up (~69 MiB) ----
  char* ws = (char*)d_ws;
  const size_t SA = 29360128; // 28 MiB activation pool (max = 2*432*496*32 f16 = 26.2 MiB)
  const size_t SM = 1048576;  // 1 MiB mask pool
  _Float16* act[2] = { (_Float16*)(ws), (_Float16*)(ws + SA) };
  _Float16* msk[2] = { (_Float16*)(ws + 2 * SA), (_Float16*)(ws + 2 * SA + SM) };
  char* wtBase = ws + 2 * SA + 2 * SM;       // 8 MiB f16 weights
  char* bnBase = wtBase + 8 * 1024 * 1024;   // folded BN params

  struct Layer { int cin, cout, stride, hin, win, hout, wout, mbuf; };
  const Layer L[14] = {
    { 32,  32, 1, 432, 496, 432, 496,  0}, // conv1[0]
    { 32,  32, 1, 432, 496, 432, 496,  0}, // conv1[1]
    { 32,  64, 2, 432, 496, 216, 248,  1}, // conv2[0]
    { 64,  64, 1, 216, 248, 216, 248,  1}, // conv2[1]
    { 64,  64, 1, 216, 248, 216, 248,  1}, // conv2[2]
    { 64, 128, 2, 216, 248, 108, 124,  0}, // conv3[0]
    {128, 128, 1, 108, 124, 108, 124,  0}, // conv3[1]
    {128, 128, 1, 108, 124, 108, 124,  0}, // conv3[2]
    {128, 256, 2, 108, 124,  54,  62,  1}, // conv4[0]
    {256, 256, 1,  54,  62,  54,  62,  1}, // conv4[1]
    {256, 256, 1,  54,  62,  54,  62,  1}, // conv4[2]
    {256, 256, 2,  54,  62,  27,  31, -1}, // conv5[0]
    {256, 256, 1,  27,  31,  27,  31, -1}, // conv5[1]
    {256, 256, 1,  27,  31,  27,  31, -1}, // conv5[2]
  };

  auto al256 = [](size_t x) { return (x + 255) & ~(size_t)255; };
  size_t wtOff[14], bnOff[14], wo = 0, bo = 0;
  for (int i = 0; i < 14; ++i) {
    wtOff[i] = wo; wo += al256((size_t)9 * L[i].cin * L[i].cout * sizeof(_Float16));
    bnOff[i] = bo; bo += al256((size_t)L[i].cout * 2 * sizeof(float));
  }

  // 1) zero x0 + full-res mask (only buffers not fully rewritten each call)
  {
    int n8 = (2 * GRID_H * GRID_W * 32) / 8;
    k_zero16<<<(n8 + 255) / 256, 256, 0, stream>>>(act[0], n8);
    int m8 = (2 * GRID_H * GRID_W) / 8;
    k_zero16<<<(m8 + 255) / 256, 256, 0, stream>>>(msk[0], m8);
  }

  // 2) scatter pillars
  {
    int tot = N * 32;
    k_scatter<<<(tot + 255) / 256, 256, 0, stream>>>(feats, coords, act[0], msk[0], N);
  }

  // 3) weight transpose->f16 and BN folding for all 14 layers
  for (int i = 0; i < 14; ++i) {
    const float* w = (const float*)d_in[3 + i * 5 + 0];
    const float* g = (const float*)d_in[3 + i * 5 + 1];
    const float* b = (const float*)d_in[3 + i * 5 + 2];
    const float* m = (const float*)d_in[3 + i * 5 + 3];
    const float* v = (const float*)d_in[3 + i * 5 + 4];
    int K = 9 * L[i].cin, C = L[i].cout;
    _Float16* wt = (_Float16*)(wtBase + wtOff[i]);
    float* sc = (float*)(bnBase + bnOff[i]);
    float* bp = sc + C;
    int tot = K * C;
    k_wt<<<(tot + 255) / 256, 256, 0, stream>>>(w, wt, K, C);
    k_bn<<<(C + 63) / 64, 64, 0, stream>>>(g, b, m, v, sc, bp, C);
  }

  // 4) conv chain with mask dilations before each stride-2 sparse stage
  int cur = 0;
  for (int i = 0; i < 14; ++i) {
    if (i == 2) { // mask s0 -> s1
      int tot = Bn * 216 * 248;
      k_dilate<<<(tot + 255) / 256, 256, 0, stream>>>(msk[0], msk[1], Bn, 432, 496, 216, 248);
    } else if (i == 5) { // s1 -> s2
      int tot = Bn * 108 * 124;
      k_dilate<<<(tot + 255) / 256, 256, 0, stream>>>(msk[1], msk[0], Bn, 216, 248, 108, 124);
    } else if (i == 8) { // s2 -> s3
      int tot = Bn * 54 * 62;
      k_dilate<<<(tot + 255) / 256, 256, 0, stream>>>(msk[0], msk[1], Bn, 108, 124, 54, 62);
    }

    const Layer& l = L[i];
    int Mt = Bn * l.hout * l.wout;
    dim3 grid((Mt + 63) / 64, l.cout / 32, 1);
    const _Float16* mp = (l.mbuf >= 0) ? msk[l.mbuf] : (const _Float16*)nullptr;
    const bool last = (i == 13);
    _Float16* wt = (_Float16*)(wtBase + wtOff[i]);
    float* sc = (float*)(bnBase + bnOff[i]);
    float* bp = sc + l.cout;
    _Float16* o16 = last ? (_Float16*)nullptr : act[1 - cur];
    float*    o32 = last ? (float*)d_out : (float*)nullptr;

    switch (l.cin) {
      case 32:
        k_conv_wmma<32><<<grid, 256, 0, stream>>>(act[cur], wt, sc, bp, mp, o16, o32,
            Bn, l.hin, l.win, l.hout, l.wout, l.cout, l.stride);
        break;
      case 64:
        k_conv_wmma<64><<<grid, 256, 0, stream>>>(act[cur], wt, sc, bp, mp, o16, o32,
            Bn, l.hin, l.win, l.hout, l.wout, l.cout, l.stride);
        break;
      case 128:
        k_conv_wmma<128><<<grid, 256, 0, stream>>>(act[cur], wt, sc, bp, mp, o16, o32,
            Bn, l.hin, l.win, l.hout, l.wout, l.cout, l.stride);
        break;
      default:
        k_conv_wmma<256><<<grid, 256, 0, stream>>>(act[cur], wt, sc, bp, mp, o16, o32,
            Bn, l.hin, l.win, l.hout, l.wout, l.cout, l.stride);
        break;
    }
    cur = 1 - cur;
  }

  (void)n_in; (void)out_size; (void)ws_size;
}
